// SubspaceHIE_29506425323606
// MI455X (gfx1250) — compile-verified, matrix operand order checked by reading, not run on hardware
//
#include <hip/hip_runtime.h>

typedef unsigned short u16;
typedef unsigned int   u32;
typedef __attribute__((ext_vector_type(16))) __bf16       bf16x16;
typedef __attribute__((ext_vector_type(8)))  float        f32x8;
typedef __attribute__((ext_vector_type(4)))  unsigned int uintx4;

union FragA { uintx4 q[2]; u16 u[16]; bf16x16 v; };
union FragC { f32x8 v; float f[8]; };

constexpr int kS = 4, kH = 1024, kD = 256, kB = 8, kN = 1024;
constexpr int kM = kB * kN;   // 8192 rows (b,n) flattened

__device__ __forceinline__ u16 f2bf(float f) {
  union { float f; unsigned u; } x; x.f = f;
  unsigned r = x.u + 0x7FFFu + ((x.u >> 16) & 1u);   // round-to-nearest-even
  return (u16)(r >> 16);
}

__device__ __forceinline__ f32x8 wmma_bf16(const FragA& a, const FragA& b, f32x8 c) {
  // D = A(16x32 bf16) * B(32x16 bf16) + C(16x16 f32)
  return __builtin_amdgcn_wmma_f32_16x16x32_bf16(false, a.v, false, b.v,
                                                 (short)0, c, false, false);
}

// Async DMA: global(16B per lane) -> LDS, tracked by ASYNCcnt (CDNA5 TDM-lite).
// Inline asm: clang builtin arity differs across toolchains (see bridge doc).
__device__ __forceinline__ void async_g2l_b128(const u16* g, void* lds) {
  u32 ldsOff = (u32)(uintptr_t)lds;   // generic LDS addr low 32 bits = LDS offset
  asm volatile("global_load_async_to_lds_b128 %0, %1, off"
               :: "v"(ldsOff), "v"(g) : "memory");
}
__device__ __forceinline__ void wait_async0() {
  asm volatile("s_wait_asynccnt 0x0" ::: "memory");
}

// ---------------------------------------------------------------- conversions
__global__ void k_cvt_bf16(const float* __restrict__ src, u16* __restrict__ dst, int n) {
  int i = blockIdx.x * blockDim.x + threadIdx.x;
  if (i < n) dst[i] = f2bf(src[i]);
}

__global__ void k_brsum(const float* __restrict__ br, float* __restrict__ out) {
  int h = blockIdx.x * blockDim.x + threadIdx.x;
  if (h < kH) {
    float s = 0.f;
    for (int ss = 0; ss < kS; ++ss) s += br[ss * kH + h];
    out[h] = s;
  }
}

// ------------------------------------------------- Stage A: projection + norms
// x_sub[s, row, d] = hx[row, :] @ Wp[s, :, d] + bp[s, d]   (bf16 out + 1/||.||)
// grid: (kM/32, kS, 2[hx|ha]); block 256 (8 waves: 2 Mx x 4 Nx), tile 32x256,K=32
__global__ __launch_bounds__(256)
void k_project(const u16* __restrict__ hxb, const u16* __restrict__ hab,
               const u16* __restrict__ Wpb, const float* __restrict__ bp,
               u16* __restrict__ xsub, u16* __restrict__ asub,
               float* __restrict__ invx, float* __restrict__ inva) {
  const int s = blockIdx.y, z = blockIdx.z;
  const int row0 = blockIdx.x * 32;
  const u16* Xsrc = z ? hab : hxb;
  u16* outp = z ? asub : xsub;
  float* invp = z ? inva : invx;

  __shared__ __align__(16) u16 At[32][32];    // A row-major
  __shared__ __align__(16) u16 Bt[256][32];   // B transposed: Bt[d][k]
  __shared__ float nrm[32];

  const int tid = threadIdx.x, lane = tid & 31, w = tid >> 5;
  const int wm = w >> 2, wn = w & 3;
  const int n = lane & 15, hh = lane >> 4;

  FragC acc[4];
#pragma unroll
  for (int t = 0; t < 4; ++t)
#pragma unroll
    for (int r = 0; r < 8; ++r) acc[t].f[r] = 0.f;

  const u16* Wps = Wpb + (size_t)s * kH * kD;

  for (int k0 = 0; k0 < kH; k0 += 32) {
    // A tile: async DMA straight into LDS (no VGPR round trip)
    if (tid < 128) {
      int mi = tid >> 2, part = tid & 3;
      const u16* gp = &Xsrc[(size_t)(row0 + mi) * kH + k0 + part * 8];
      async_g2l_b128(gp, &At[mi][part * 8]);
      if (k0 + 32 < kH)
        __builtin_prefetch(&Xsrc[(size_t)(row0 + mi) * kH + k0 + 32], 0, 0);
    }
    // B tile: through VGPRs (feeds the in-register transpose scatter)
#pragma unroll
    for (int rep = 0; rep < 4; ++rep) {
      int g = rep * 256 + tid;          // 1024 groups of 8 halves (32x256)
      int k = g >> 5, dc = g & 31;
      uintx4 vq = *(const uintx4*)&Wps[(size_t)(k0 + k) * kD + dc * 8];
      const u16* vh = (const u16*)&vq;
#pragma unroll
      for (int x = 0; x < 8; ++x) Bt[dc * 8 + x][k] = vh[x];
    }
    wait_async0();
    __syncthreads();

    FragA af;
    af.q[0] = *(const uintx4*)&At[wm * 16 + n][hh * 8];
    af.q[1] = *(const uintx4*)&At[wm * 16 + n][16 + hh * 8];
#pragma unroll
    for (int t = 0; t < 4; ++t) {
      FragA bf;
      int col = wn * 64 + t * 16 + n;
      bf.q[0] = *(const uintx4*)&Bt[col][hh * 8];
      bf.q[1] = *(const uintx4*)&Bt[col][16 + hh * 8];
      acc[t].v = wmma_bf16(af, bf, acc[t].v);
    }
    __syncthreads();
  }

  if (tid < 32) nrm[tid] = 0.f;
  __syncthreads();
#pragma unroll
  for (int t = 0; t < 4; ++t) {
    int col = wn * 64 + t * 16 + n;
    float bv = bp[s * kD + col];
#pragma unroll
    for (int r = 0; r < 8; ++r) {
      acc[t].f[r] += bv;
      atomicAdd(&nrm[wm * 16 + r + 8 * hh], acc[t].f[r] * acc[t].f[r]);
    }
  }
  __syncthreads();
  if (tid < 32)
    invp[(size_t)s * kM + row0 + tid] = 1.f / fmaxf(sqrtf(nrm[tid]), 1e-8f);
#pragma unroll
  for (int t = 0; t < 4; ++t) {
    int col = wn * 64 + t * 16 + n;
#pragma unroll
    for (int r = 0; r < 8; ++r) {
      int row = row0 + wm * 16 + r + 8 * hh;
      outp[((size_t)s * kM + row) * kD + col] = f2bf(acc[t].f[r]);
    }
  }
}

// --------------------------- Stage B: fused sim -> softmax(S) -> weighted agg
// grid: kB*(kN/16); block 256 = 8 waves (s = w&3, d-half t = w>>2).
// sim/w never leave LDS; weighted accumulated in VGPRs over all j.
__global__ __launch_bounds__(256)
void k_attend(const u16* __restrict__ xsub, const u16* __restrict__ asub,
              const float* __restrict__ invx, const float* __restrict__ inva,
              u16* __restrict__ wout) {                 // [kM][kS*kD]
  const int nb = kN / 16;
  const int b = blockIdx.x / nb;
  const int i0 = (blockIdx.x % nb) * 16;

  __shared__ __align__(16) u16 aS[kS][16][kD];   // a_sub j-tile, row-major   32KB
  __shared__ __align__(16) u16 aT[kS][kD][16];   // transposed for weighted   32KB
  __shared__ float simP[2][kS][16][16];          // partial sims (d-halves)    8KB
  __shared__ __align__(16) u16 wS[kS][16][16];   // softmax weights (bf16)     2KB
  __shared__ float invAj[kS][16];

  const int tid = threadIdx.x, lane = tid & 31, w = tid >> 5;
  const int s = w & 3, t = w >> 2;
  const int n = lane & 15, hh = lane >> 4;

  // Preload this wave's x fragments (16 rows x its 128-wide d-half), reused 64x
  FragA xf[4];
  {
    const u16* xp = xsub + ((size_t)s * kM + (size_t)b * kN + i0 + n) * kD;
#pragma unroll
    for (int kd4 = 0; kd4 < 4; ++kd4) {
      int kd = t * 128 + kd4 * 32;
      xf[kd4].q[0] = *(const uintx4*)&xp[kd + hh * 8];
      xf[kd4].q[1] = *(const uintx4*)&xp[kd + 16 + hh * 8];
    }
  }

  // softmax worker coords: thread (si, sj) element of the 16x16 sim tile
  const int si = tid >> 4, sj = tid & 15;
  float ivx[kS];
#pragma unroll
  for (int s4 = 0; s4 < kS; ++s4)
    ivx[s4] = invx[(size_t)s4 * kM + (size_t)b * kN + i0 + si];

  FragC acc[8];
#pragma unroll
  for (int q = 0; q < 8; ++q)
#pragma unroll
    for (int r = 0; r < 8; ++r) acc[q].f[r] = 0.f;

  for (int j0 = 0; j0 < kN; j0 += 16) {
    // cooperative load of a_sub tile for all 4 subspaces (+ transpose copy)
#pragma unroll
    for (int rep = 0; rep < 8; ++rep) {
      int g = rep * 256 + tid;          // 2048 groups of 8 halves
      int s_i = g >> 9, rem = g & 511;
      int jj = rem >> 5, dc = rem & 31;
      uintx4 vq = *(const uintx4*)
          &asub[((size_t)s_i * kM + (size_t)b * kN + j0 + jj) * kD + dc * 8];
      *(uintx4*)&aS[s_i][jj][dc * 8] = vq;
      const u16* vh = (const u16*)&vq;
#pragma unroll
      for (int x = 0; x < 8; ++x) aT[s_i][dc * 8 + x][jj] = vh[x];
    }
    if (tid < 64) {
      int ss = tid >> 4, jj = tid & 15;
      invAj[ss][jj] = inva[(size_t)ss * kM + (size_t)b * kN + j0 + jj];
    }
    __syncthreads();

    // partial sim for (s, d-half t): 4 WMMA over K=128
    FragC sacc;
#pragma unroll
    for (int r = 0; r < 8; ++r) sacc.f[r] = 0.f;
#pragma unroll
    for (int kd4 = 0; kd4 < 4; ++kd4) {
      int kd = t * 128 + kd4 * 32;
      FragA bfr;
      bfr.q[0] = *(const uintx4*)&aS[s][n][kd + hh * 8];
      bfr.q[1] = *(const uintx4*)&aS[s][n][kd + 16 + hh * 8];
      sacc.v = wmma_bf16(xf[kd4], bfr, sacc.v);
    }
#pragma unroll
    for (int r = 0; r < 8; ++r) simP[t][s][r + 8 * hh][n] = sacc.f[r];
    __syncthreads();

    // elementwise 4-way softmax across subspaces (cosine scaling folded in)
    {
      float v[kS], mx = -3.4e38f;
#pragma unroll
      for (int s4 = 0; s4 < kS; ++s4) {
        v[s4] = (simP[0][s4][si][sj] + simP[1][s4][si][sj]) * ivx[s4] * invAj[s4][sj];
        mx = fmaxf(mx, v[s4]);
      }
      float sum = 0.f;
#pragma unroll
      for (int s4 = 0; s4 < kS; ++s4) { v[s4] = __expf(v[s4] - mx); sum += v[s4]; }
      float rs = 1.f / sum;
#pragma unroll
      for (int s4 = 0; s4 < kS; ++s4) wS[s4][si][sj] = f2bf(v[s4] * rs);
    }
    __syncthreads();

    // weighted[i, d] += w(16x16, K zero-padded to 32) @ a_sub(j-tile, d-half)
    FragA wf;
    wf.q[0] = *(const uintx4*)&wS[s][n][hh * 8];
    uintx4 zz = {0u, 0u, 0u, 0u};
    wf.q[1] = zz;
#pragma unroll
    for (int nt = 0; nt < 8; ++nt) {
      FragA bfr;
      int d0 = t * 128 + nt * 16;
      bfr.q[0] = *(const uintx4*)&aT[s][d0 + n][hh * 8];
      bfr.q[1] = zz;
      acc[nt].v = wmma_bf16(wf, bfr, acc[nt].v);
    }
    __syncthreads();
  }

  // store weighted as [row][(s,d)] bf16 — Stage C A-operand layout
#pragma unroll
  for (int nt = 0; nt < 8; ++nt) {
    int col = s * kD + t * 128 + nt * 16 + n;
#pragma unroll
    for (int r = 0; r < 8; ++r) {
      int row = b * kN + i0 + r + 8 * hh;
      wout[(size_t)row * (kS * kD) + col] = f2bf(acc[nt].f[r]);
    }
  }
}

// ------------------------ Stage C: out = weighted @ Wr (K = S*D=1024) + br_sum
// grid: (kM/64, kH/64); block 256 (8 waves: 4 Mx x 2 Nx), tile 64x64, K=32
__global__ __launch_bounds__(256)
void k_reconstruct(const u16* __restrict__ wgt, const u16* __restrict__ Wrb,
                   const float* __restrict__ brs, float* __restrict__ out) {
  const int row0 = blockIdx.x * 64;
  const int n0 = blockIdx.y * 64;
  const int K = kS * kD;

  __shared__ __align__(16) u16 At[64][32];
  __shared__ __align__(16) u16 Bt[64][32];   // Bt[n][k]

  const int tid = threadIdx.x, lane = tid & 31, w = tid >> 5;
  const int wm = w >> 1, wn = w & 1;
  const int n = lane & 15, hh = lane >> 4;

  FragC acc[2];
#pragma unroll
  for (int t = 0; t < 2; ++t)
#pragma unroll
    for (int r = 0; r < 8; ++r) acc[t].f[r] = 0.f;

  for (int k0 = 0; k0 < K; k0 += 32) {
    // A tile: async DMA into LDS
    {
      int mi = tid >> 2, part = tid & 3;
      const u16* gp = &wgt[(size_t)(row0 + mi) * K + k0 + part * 8];
      async_g2l_b128(gp, &At[mi][part * 8]);
      if (k0 + 32 < K)
        __builtin_prefetch(&wgt[(size_t)(row0 + mi) * K + k0 + 32], 0, 0);
    }
    // B tile transposed: through VGPRs
    {
      int k = tid >> 3, nc = tid & 7;
      uintx4 vq = *(const uintx4*)&Wrb[(size_t)(k0 + k) * kH + n0 + nc * 8];
      const u16* vh = (const u16*)&vq;
#pragma unroll
      for (int x = 0; x < 8; ++x) Bt[nc * 8 + x][k] = vh[x];
    }
    wait_async0();
    __syncthreads();

    FragA af;
    af.q[0] = *(const uintx4*)&At[wm * 16 + n][hh * 8];
    af.q[1] = *(const uintx4*)&At[wm * 16 + n][16 + hh * 8];
#pragma unroll
    for (int t = 0; t < 2; ++t) {
      FragA bfr;
      int col = wn * 32 + t * 16 + n;
      bfr.q[0] = *(const uintx4*)&Bt[col][hh * 8];
      bfr.q[1] = *(const uintx4*)&Bt[col][16 + hh * 8];
      acc[t].v = wmma_bf16(af, bfr, acc[t].v);
    }
    __syncthreads();
  }

#pragma unroll
  for (int t = 0; t < 2; ++t) {
    int col = n0 + wn * 32 + t * 16 + n;
    float bv = brs[col];
#pragma unroll
    for (int r = 0; r < 8; ++r) {
      int row = row0 + wm * 16 + r + 8 * hh;
      out[(size_t)row * kH + col] = acc[t].f[r] + bv;
    }
  }
}

// ---------------------------------------------------------------------- launch
extern "C" void kernel_launch(void* const* d_in, const int* in_sizes, int n_in,
                              void* d_out, int out_size, void* d_ws, size_t ws_size,
                              hipStream_t stream) {
  (void)in_sizes; (void)n_in; (void)out_size; (void)ws_size;
  const float* hx = (const float*)d_in[0];
  const float* ha = (const float*)d_in[1];
  const float* Wp = (const float*)d_in[2];
  const float* bp = (const float*)d_in[3];
  const float* Wr = (const float*)d_in[4];
  const float* br = (const float*)d_in[5];
  float* out = (float*)d_out;

  char* p = (char*)d_ws;
  size_t off = 0;
  auto take = [&](size_t bytes) -> void* {
    void* r = p + off;
    off += (bytes + 255) & ~(size_t)255;
    return r;
  };
  u16* hxb   = (u16*)take((size_t)kM * kH * 2);
  u16* hab   = (u16*)take((size_t)kM * kH * 2);
  u16* Wpb   = (u16*)take((size_t)kS * kH * kD * 2);
  u16* Wrb   = (u16*)take((size_t)kS * kD * kH * 2);
  float* brs = (float*)take((size_t)kH * 4);
  u16* xsub  = (u16*)take((size_t)kS * kM * kD * 2);
  u16* asub  = (u16*)take((size_t)kS * kM * kD * 2);
  float* ivx = (float*)take((size_t)kS * kM * 4);
  float* iva = (float*)take((size_t)kS * kM * 4);
  u16* wgt   = (u16*)take((size_t)kM * kS * kD * 2);

  int nhx = kM * kH;
  k_cvt_bf16<<<(nhx + 255) / 256, 256, 0, stream>>>(hx, hxb, nhx);
  k_cvt_bf16<<<(nhx + 255) / 256, 256, 0, stream>>>(ha, hab, nhx);
  int nw = kS * kH * kD;
  k_cvt_bf16<<<(nw + 255) / 256, 256, 0, stream>>>(Wp, Wpb, nw);
  k_cvt_bf16<<<(nw + 255) / 256, 256, 0, stream>>>(Wr, Wrb, nw);
  k_brsum<<<(kH + 255) / 256, 256, 0, stream>>>(br, brs);

  dim3 gA(kM / 32, kS, 2);
  k_project<<<gA, 256, 0, stream>>>(hxb, hab, Wpb, bp, xsub, asub, ivx, iva);

  k_attend<<<kB * (kN / 16), 256, 0, stream>>>(xsub, asub, ivx, iva, wgt);

  dim3 gC(kM / 64, kH / 64);
  k_reconstruct<<<gC, 256, 0, stream>>>(wgt, Wrb, brs, out);
}
